// MoEDownsample_6322191860389
// MI455X (gfx1250) — matrix operand order repaired and checked by reading, not compiled, not run on hardware
//
#include <hip/hip_runtime.h>
#include <hip/hip_fp16.h>
#include <math.h>

typedef __attribute__((ext_vector_type(16))) _Float16 v16h;
typedef __attribute__((ext_vector_type(8)))  _Float16 v8h;
typedef __attribute__((ext_vector_type(8)))  float    v8f;

union Frag { v16h v; v8h h[2]; };

// ---------------------------------------------------------------------------
// Kernel 1: global average pool  pooled[b*64+ci] = mean(x[b,ci,:,:])
// ---------------------------------------------------------------------------
__global__ __launch_bounds__(256)
void moe_pool_kernel(const float* __restrict__ x, float* __restrict__ pooled) {
    __shared__ float red[256];
    const int bc = blockIdx.x;                 // b*64 + ci, 0..1023
    const float* p = x + (size_t)bc * 65536;   // 256*256
    float s = 0.f;
    for (int i = threadIdx.x; i < 65536; i += 256) s += p[i];
    red[threadIdx.x] = s;
    __syncthreads();
    for (int off = 128; off > 0; off >>= 1) {
        if (threadIdx.x < off) red[threadIdx.x] += red[threadIdx.x + off];
        __syncthreads();
    }
    if (threadIdx.x == 0) pooled[bc] = red[0] * (1.f / 65536.f);
}

// ---------------------------------------------------------------------------
// Kernel 2: gate = softmax(pooled @ gw^T + gb) -> top2 renormalized
// ---------------------------------------------------------------------------
__global__ void moe_gate_kernel(const float* __restrict__ pooled,
                                const float* __restrict__ gw,
                                const float* __restrict__ gb,
                                int* __restrict__ gidx, float* __restrict__ gwt) {
    const int b = threadIdx.x;
    if (b >= 16) return;
    float logit[4];
    for (int e = 0; e < 4; ++e) {
        float s = gb[e];
        for (int c = 0; c < 64; ++c) s += pooled[b * 64 + c] * gw[e * 64 + c];
        logit[e] = s;
    }
    float m = fmaxf(fmaxf(logit[0], logit[1]), fmaxf(logit[2], logit[3]));
    float g[4], sum = 0.f;
    for (int e = 0; e < 4; ++e) { g[e] = __expf(logit[e] - m); sum += g[e]; }
    for (int e = 0; e < 4; ++e) g[e] /= sum;
    int i0 = 0;
    for (int e = 1; e < 4; ++e) if (g[e] > g[i0]) i0 = e;
    int i1 = -1;
    for (int e = 0; e < 4; ++e) { if (e == i0) continue; if (i1 < 0 || g[e] > g[i1]) i1 = e; }
    float wsum = g[i0] + g[i1] + 1e-8f;
    gidx[b * 2 + 0] = i0;          gidx[b * 2 + 1] = i1;
    gwt[b * 2 + 0] = g[i0] / wsum; gwt[b * 2 + 1] = g[i1] / wsum;
}

// ---------------------------------------------------------------------------
// Kernel 3: implicit-GEMM dilated stride-2 conv via WMMA f16 -> f32,
// fused bias + BN + exact GELU + gate weight, for the selected expert only.
// Block = (b, slot, ho): M=64 out-channels x N=128 wo positions.
// K-index = ((kh*k)+kw)*64 + ci  (so each 32-chunk has fixed kh,kw).
// (kh,kw) is tracked incrementally across chunks -> no integer division in
// the K loop (rr advances only on odd->even chunk transitions).
// ---------------------------------------------------------------------------
__global__ __launch_bounds__(256)
void moe_conv_kernel(const float* __restrict__ x,
                     const float* __restrict__ w0, const float* __restrict__ w1,
                     const float* __restrict__ w2, const float* __restrict__ w3,
                     const float* __restrict__ b0, const float* __restrict__ b1,
                     const float* __restrict__ b2, const float* __restrict__ b3,
                     const float* __restrict__ bn_scale, const float* __restrict__ bn_bias,
                     const float* __restrict__ bn_mean,  const float* __restrict__ bn_var,
                     const int* __restrict__ gidx, const float* __restrict__ gwt,
                     float* __restrict__ out) {
    __shared__ _Float16 a_lds[2][64][40];     // [buf][m][kk]   (padded row)
    __shared__ _Float16 b_lds[2][128][40];    // [buf][n][kk]   (padded row)
    __shared__ float alpha_s[64], beta_s[64];

    const int ho   = blockIdx.x;   // 0..127
    const int slot = blockIdx.y;   // 0..1
    const int b    = blockIdx.z;   // 0..15
    const int tid  = threadIdx.x;

    const int   e     = gidx[b * 2 + slot];
    const float slotw = gwt[b * 2 + slot];

    const int kKs[4] = {3, 5, 7, 9};
    const int dls[4] = {1, 2, 3, 4};
    const int kK   = kKs[e];
    const int dil  = dls[e];
    const int pad  = dil * (kK - 1) / 2;
    const int kKsq = kK * kK;
    const int nK   = (64 * kKsq) >> 5;        // chunks of K=32; always exact

    const float* wsrc = (e == 0) ? w0 : (e == 1) ? w1 : (e == 2) ? w2 : w3;
    const float* bsrc = (e == 0) ? b0 : (e == 1) ? b1 : (e == 2) ? b2 : b3;

    if (tid < 64) {
        const int m = tid;
        float inv = bn_scale[e * 64 + m] * rsqrtf(bn_var[e * 64 + m] + 1e-5f);
        alpha_s[m] = inv;
        beta_s[m]  = bn_bias[e * 64 + m] - bn_mean[e * 64 + m] * inv + bsrc[m] * inv;
    }

    // loader thread mapping
    const int am  = tid >> 2;      // A: out-channel 0..63
    const int ag  = tid & 3;       // A: k-subgroup of 8
    const int bk  = tid >> 3;      // B: kk 0..31
    const int bg  = tid & 7;       // B: n-group of 16

    const int aBase = am * 64 * kKsq;          // weight row base for this thread
    const int hoBase = 2 * ho - pad;

    // chunk c params: cib = (c&1)*32; rr = c>>1; kh = rr/kK; kw = rr%kK
    auto load_chunk = [&](int buf, int cib, int kh, int kw) {
        // ---- A tile: 64 x 32 weights, f32 -> f16 ----
        {
            const int khkw = kh * kK + kw;
            const int kk0  = ag * 8;
            #pragma unroll
            for (int j = 0; j < 8; ++j) {
                const int kk = kk0 + j;
                float v = wsrc[aBase + (cib + kk) * kKsq + khkw];
                a_lds[buf][am][kk] = (_Float16)v;
            }
        }
        // ---- B tile: 32(K) x 128(N) im2col, stored [n][kk] ----
        {
            const int ci = cib + bk;
            const int hi = hoBase + kh * dil;
            const bool hok = (hi >= 0) && (hi < 256);
            const float* src = x + (((size_t)(b * 64 + ci) * 256 + (hok ? hi : 0)) << 8);
            const int wofs = kw * dil - pad;
            const int n0 = bg * 16;
            #pragma unroll
            for (int j = 0; j < 16; ++j) {
                const int n  = n0 + j;
                const int wi = 2 * n + wofs;
                float v = (hok && wi >= 0 && wi < 256) ? src[wi] : 0.f;
                b_lds[buf][n][bk] = (_Float16)v;
            }
        }
    };

    // wave tiling: 8 waves -> (wm 0..3) x (wn 0..1); each wave = 16M x 64N
    const int wave = tid >> 5, lane = tid & 31;
    const int wm = wave & 3, wn = wave >> 2;
    const int l   = lane & 15;
    const int kbA = (lane < 16) ? 0 : 8;    // A layout: K 0-7/16-23 vs 8-15/24-31
    const int kbB = (lane < 16) ? 0 : 8;    // B layout: K 0-7/16-23 vs 8-15/24-31

    v8f acc[4] = {v8f{}, v8f{}, v8f{}, v8f{}};

    int kh = 0, kw = 0;                 // params of the *next* chunk to load
    load_chunk(0, 0, 0, 0);             // chunk 0: cib=0, kh=0, kw=0
    __syncthreads();

    for (int kc = 0; kc < nK; ++kc) {
        const int buf = kc & 1;

        // advance (kh,kw) to chunk kc+1: rr changes only when kc is odd
        if (kc & 1) { if (++kw == kK) { kw = 0; ++kh; } }
        if (kc + 1 < nK) load_chunk(buf ^ 1, ((kc + 1) & 1) << 5, kh, kw);

        Frag af;
        const _Float16* ap = &a_lds[buf][wm * 16 + l][0];
        af.h[0] = *(const v8h*)(ap + kbA);
        af.h[1] = *(const v8h*)(ap + 16 + kbA);
        #pragma unroll
        for (int nt = 0; nt < 4; ++nt) {
            Frag bf;
            const _Float16* bp = &b_lds[buf][wn * 64 + nt * 16 + l][kbB];
            bf.h[0] = *(const v8h*)bp;
            bf.h[1] = *(const v8h*)(bp + 16);
            acc[nt] = __builtin_amdgcn_wmma_f32_16x16x32_f16(
                false, af.v, false, bf.v, (short)0, acc[nt], false, false);
        }
        __syncthreads();
    }

    // epilogue: bias+BN folded, exact GELU, gate weight
    const int mbase = wm * 16 + ((lane < 16) ? 0 : 8);
    const int nlow  = lane & 15;
    float* obase = out + (((size_t)(b * 128 + slot * 64)) * 128 + ho) * 128;
    #pragma unroll
    for (int nt = 0; nt < 4; ++nt) {
        const int n = wn * 64 + nt * 16 + nlow;
        #pragma unroll
        for (int r = 0; r < 8; ++r) {
            const int m = mbase + r;
            float v = acc[nt][r] * alpha_s[m] + beta_s[m];
            float g = 0.5f * v * (1.0f + erff(v * 0.70710678118654752f));
            obase[(size_t)m * 16384 + n] = g * slotw;
        }
    }
}

// ---------------------------------------------------------------------------
extern "C" void kernel_launch(void* const* d_in, const int* in_sizes, int n_in,
                              void* d_out, int out_size, void* d_ws, size_t ws_size,
                              hipStream_t stream) {
    const float* x  = (const float*)d_in[0];
    const float* w0 = (const float*)d_in[1];
    const float* b0 = (const float*)d_in[2];
    const float* w1 = (const float*)d_in[3];
    const float* b1 = (const float*)d_in[4];
    const float* w2 = (const float*)d_in[5];
    const float* b2 = (const float*)d_in[6];
    const float* w3 = (const float*)d_in[7];
    const float* b3 = (const float*)d_in[8];
    const float* bn_scale = (const float*)d_in[9];
    const float* bn_bias  = (const float*)d_in[10];
    const float* bn_mean  = (const float*)d_in[11];
    const float* bn_var   = (const float*)d_in[12];
    const float* gate_w   = (const float*)d_in[13];
    const float* gate_b   = (const float*)d_in[14];
    float* out = (float*)d_out;

    // workspace layout (tiny): pooled[1024] f32 | gidx[32] i32 | gwt[32] f32
    float* pooled = (float*)d_ws;
    int*   gidx   = (int*)((char*)d_ws + 4096);
    float* gwt    = (float*)((char*)d_ws + 4096 + 128);

    moe_pool_kernel<<<1024, 256, 0, stream>>>(x, pooled);
    moe_gate_kernel<<<1, 32, 0, stream>>>(pooled, gate_w, gate_b, gidx, gwt);

    dim3 grid(128, 2, 16);   // (ho, slot, b)
    moe_conv_kernel<<<grid, 256, 0, stream>>>(
        x, w0, w1, w2, w3, b0, b1, b2, b3,
        bn_scale, bn_bias, bn_mean, bn_var, gidx, gwt, out);
}
